// memory_73143293051046
// MI455X (gfx1250) — matrix operand order repaired and checked by reading, not compiled, not run on hardware
//
#include <hip/hip_runtime.h>

// ---------------- problem constants ----------------
constexpr int B_   = 4;
constexpr int T_   = 4096;
constexpr int DIM_ = 1024;
constexpr int S_   = 8;
constexpr int H_   = 8;
constexpr int HD_  = 128;           // DIM / H
constexpr int L_   = T_ / S_;       // 512

constexpr int GM = B_ * T_;         // 16384
constexpr int GN = DIM_;            // 1024
constexpr int GK = DIM_;            // 1024

// ---------------- types ----------------
typedef __attribute__((ext_vector_type(16))) __bf16 v16bf;
typedef __attribute__((ext_vector_type(8)))  __bf16 v8bf;
typedef __attribute__((ext_vector_type(8)))  float  v8f;
typedef __attribute__((ext_vector_type(4)))  float  v4f;
typedef __attribute__((ext_vector_type(4)))  int    v4i;

#define ASG __attribute__((address_space(1)))
#define ASL __attribute__((address_space(3)))

#if defined(__gfx1250__) && __has_builtin(__builtin_amdgcn_global_load_async_to_lds_b128)
#define HAVE_ASYNC_LDS 1
#else
#define HAVE_ASYNC_LDS 0
#endif

static __device__ __forceinline__ void wait_async0() {
#if __has_builtin(__builtin_amdgcn_s_wait_asynccnt)
  __builtin_amdgcn_s_wait_asynccnt(0);
#else
  asm volatile("s_wait_asynccnt 0" ::: "memory");
#endif
}

// native hw converts (RNE) instead of manual bit-twiddling
static __device__ __forceinline__ __bf16 f2bf(float f)  { return (__bf16)f; }
static __device__ __forceinline__ float  bf2f(__bf16 h) { return (float)h; }

// ---------------- 0: f32 -> bf16 cast, 8 elems/thread (b128 in, b128 out) ----
__global__ __launch_bounds__(256) void cast_f32_bf16(const float* __restrict__ in,
                                                     __bf16* __restrict__ out, int n8) {
  int i = blockIdx.x * 256 + threadIdx.x;
  if (i >= n8) return;
  const v4f* p = (const v4f*)in + (size_t)i * 2;
  v4f a = p[0], b = p[1];
  union { v8bf v; __bf16 e[8]; } u;
#pragma unroll
  for (int j = 0; j < 4; ++j) { u.e[j] = f2bf(a[j]); u.e[4 + j] = f2bf(b[j]); }
  ((v8bf*)out)[i] = u.v;
}

// ---------------- 1: big WMMA GEMM  C(M,N) = A(M,K) x W(N,K)^T ----------------
// 64x64 tile per wave = 4x4 v_wmma_f32_16x16x32_bf16 fragments.
// Fragment element maps follow ISA 7.12.2 (16-bit A 16x32 / B 32x16 / f32 C 16x16).
__global__ __launch_bounds__(256) void wmma_gemm_xw(const __bf16* __restrict__ A,
                                                    const __bf16* __restrict__ W,
                                                    __bf16* __restrict__ C) {
  const int lane = threadIdx.x & 31;
  const int wave = threadIdx.x >> 5;
  const int gw   = blockIdx.x * 8 + wave;      // 4096 wave tiles
  const int tilesN = GN / 64;                  // 16
  const int tm = (gw / tilesN) * 64;
  const int tn = (gw % tilesN) * 64;

  v8f acc[4][4] = {};

  // A: lanes 0-15 -> row m=lane, k chunks {0..7,16..23}; lanes 16-31 -> {8..15,24..31}
  const int aRow = tm + (lane & 15);
  const int aSel = (lane < 16) ? 0 : 8;
  // B: lane n = lane%16; k chunks {0..7,16..23} (lanes<16) / {8..15,24..31} (lanes>=16)
  const int bCol = tn + (lane & 15);
  const int bSel = (lane < 16) ? 0 : 8;

  const __bf16* Abase = A + (size_t)aRow * GK + aSel;
  const __bf16* Wbase = W + (size_t)bCol * GK + bSel;

  for (int kc = 0; kc < GK; kc += 32) {
    __builtin_prefetch(Abase + kc + 32, 0, 1);   // speculative next k-slice

    v16bf afrag[4];
#pragma unroll
    for (int mi = 0; mi < 4; ++mi) {
      const __bf16* p = Abase + (size_t)mi * 16 * GK + kc;
      union { v16bf v; v8bf h[2]; } u;
      u.h[0] = *(const v8bf*)(p);        // k = base .. base+7
      u.h[1] = *(const v8bf*)(p + 16);   // k = base+16 .. base+23
      afrag[mi] = u.v;
    }
    v16bf bfrag[4];
#pragma unroll
    for (int ni = 0; ni < 4; ++ni) {
      const __bf16* p = Wbase + (size_t)ni * 16 * GK + kc;
      union { v16bf v; v8bf h[2]; } u;
      u.h[0] = *(const v8bf*)(p);        // k = kb .. kb+7
      u.h[1] = *(const v8bf*)(p + 16);   // k = kb+16 .. kb+23
      bfrag[ni] = u.v;
    }
#pragma unroll
    for (int mi = 0; mi < 4; ++mi)
#pragma unroll
      for (int ni = 0; ni < 4; ++ni)
        acc[mi][ni] = __builtin_amdgcn_wmma_f32_16x16x32_bf16(
            false, afrag[mi], false, bfrag[ni], (short)0, acc[mi][ni], false, false);
  }

  // C/D layout: lane n = lane%16; VGPR i holds row m = 8*(lane>=16) + i
  const int cn = tn + (lane & 15);
  const int mB = (lane < 16) ? 0 : 8;
#pragma unroll
  for (int mi = 0; mi < 4; ++mi)
#pragma unroll
    for (int ni = 0; ni < 4; ++ni)
#pragma unroll
      for (int i = 0; i < 8; ++i)
        C[(size_t)(tm + mi * 16 + mB + i) * GN + cn + ni * 16] = f2bf(acc[mi][ni][i]);
}

// ---------------- 2: tiny GEMM  qw(S,DIM) = memory_slots(S,DIM) x Wq(DIM,DIM)^T --
__global__ __launch_bounds__(256) void qw_gemm(const float* __restrict__ ms,
                                               const float* __restrict__ Wq,
                                               float* __restrict__ qw) {
  int idx = blockIdx.x * 256 + threadIdx.x;       // 8192
  int s = idx >> 10, j = idx & 1023;
  const float* a = ms + s * DIM_;
  const float* w = Wq + (size_t)j * DIM_;
  float acc = 0.f;
  for (int k = 0; k < DIM_; ++k) acc += a[k] * w[k];
  qw[idx] = acc;
}

// ---------------- 3: write attention per (b,s,h) ----------------
__global__ __launch_bounds__(128) void write_attn(const float* __restrict__ qw,
                                                  const __bf16* __restrict__ kb,
                                                  const __bf16* __restrict__ vb,
                                                  float* __restrict__ memraw) {
  const int h = blockIdx.x % H_;
  const int s = (blockIdx.x / H_) % S_;
  const int b = blockIdx.x / (H_ * S_);
  const int tid = threadIdx.x;

  __shared__ float qs[HD_];
  __shared__ float sc[L_];
  __shared__ float red[128];

  if (tid < HD_) qs[tid] = qw[s * DIM_ + h * HD_ + tid];
  __syncthreads();

  const size_t rowBase = ((size_t)b * T_ + (size_t)s * L_) * DIM_ + h * HD_;
  const float scale = rsqrtf((float)HD_);
  for (int l = tid; l < L_; l += 128) {
    const __bf16* kp = kb + rowBase + (size_t)l * DIM_;
    float acc = 0.f;
    for (int d = 0; d < HD_; ++d) acc += qs[d] * bf2f(kp[d]);
    sc[l] = acc * scale;
  }
  __syncthreads();

  float lm = -3.4e38f;
  for (int l = tid; l < L_; l += 128) lm = fmaxf(lm, sc[l]);
  red[tid] = lm; __syncthreads();
  for (int o = 64; o > 0; o >>= 1) { if (tid < o) red[tid] = fmaxf(red[tid], red[tid + o]); __syncthreads(); }
  const float mx = red[0]; __syncthreads();

  float ls = 0.f;
  for (int l = tid; l < L_; l += 128) { float e = __expf(sc[l] - mx); sc[l] = e; ls += e; }
  red[tid] = ls; __syncthreads();
  for (int o = 64; o > 0; o >>= 1) { if (tid < o) red[tid] += red[tid + o]; __syncthreads(); }
  const float inv = 1.f / red[0]; __syncthreads();

  // tid == d (HD_ == blockDim)
  float acc = 0.f;
  const __bf16* vp = vb + rowBase + tid;
  for (int l = 0; l < L_; ++l) acc += sc[l] * bf2f(vp[(size_t)l * DIM_]);
  memraw[((size_t)b * S_ + s) * DIM_ + h * HD_ + tid] = acc * inv;
}

// ---------------- 4: rms-norm -> memory (into d_out tail) ----------------
__global__ __launch_bounds__(256) void mem_norm(const float* __restrict__ memraw,
                                                const float* __restrict__ wm,
                                                float* __restrict__ memory) {
  const int bs = blockIdx.x;               // 0..31
  const int tid = threadIdx.x;
  __shared__ float red[256];
  const float* mr = memraw + (size_t)bs * DIM_;
  float vals[4], ss = 0.f;
#pragma unroll
  for (int i = 0; i < 4; ++i) {
    int d = tid + i * 256;
    float y = mr[d] * wm[d];
    vals[i] = y; ss += y * y;
  }
  red[tid] = ss; __syncthreads();
  for (int o = 128; o > 0; o >>= 1) { if (tid < o) red[tid] += red[tid + o]; __syncthreads(); }
  const float r = rsqrtf(red[0] / (float)DIM_ + 1e-6f);
#pragma unroll
  for (int i = 0; i < 4; ++i) memory[(size_t)bs * DIM_ + tid + i * 256] = vals[i] * r;
}

// ---------------- 5: kv = memory(32,DIM) x W_read_kv(2DIM,DIM)^T ----------------
__global__ __launch_bounds__(256) void read_kv(const float* __restrict__ memory,
                                               const float* __restrict__ Wkv,
                                               float* __restrict__ kr,
                                               float* __restrict__ vr) {
  int idx = blockIdx.x * 256 + threadIdx.x;     // 32 * 2048 = 65536
  int bs = idx >> 11;
  int j  = idx & 2047;
  const float* m = memory + (size_t)bs * DIM_;
  const float* w = Wkv + (size_t)j * DIM_;
  float acc = 0.f;
  for (int k = 0; k < DIM_; ++k) acc += m[k] * w[k];
  if (j < DIM_) kr[(size_t)bs * DIM_ + j] = acc;
  else          vr[(size_t)bs * DIM_ + (j - DIM_)] = acc;
}

// ---------------- 6: read attention + SiLU per (b,t) ----------------
// kr panel (8 x 1024 f32 = 32KB, reused by 8 dots x 128 threads) is staged
// into LDS with CDNA5 async loads when the toolchain exposes them.
__global__ __launch_bounds__(128) void read_attn(const __bf16* __restrict__ qrb,
                                                 const float* __restrict__ kr,
                                                 const float* __restrict__ vr,
                                                 float* __restrict__ out) {
  const int bt = blockIdx.x;                 // 16384
  const int b = bt / T_;
  const int t = bt % T_;
  const int tid = threadIdx.x;
  const int seg = t / L_;

  __shared__ float qs[DIM_];
  __shared__ float krs[S_ * DIM_];
  __shared__ float red[128];
  __shared__ float scores[S_];

  const float* krb = kr + (size_t)b * S_ * DIM_;
#if HAVE_ASYNC_LDS
  for (int c = tid; c < (S_ * DIM_) / 4; c += 128) {
    __builtin_amdgcn_global_load_async_to_lds_b128(
        (ASG v4i*)(krb + c * 4), (ASL v4i*)(&krs[c * 4]), 0, 0);
  }
#else
  for (int c = tid; c < S_ * DIM_; c += 128) krs[c] = krb[c];
#endif

  const __bf16* qp = qrb + (size_t)bt * DIM_;
  for (int d = tid; d < DIM_; d += 128) qs[d] = bf2f(qp[d]);

#if HAVE_ASYNC_LDS
  wait_async0();
#endif
  __syncthreads();

  for (int s = 0; s < S_; ++s) {
    const float* krow = &krs[s * DIM_];
    float p = 0.f;
    for (int d = tid; d < DIM_; d += 128) p += qs[d] * krow[d];
    red[tid] = p; __syncthreads();
    for (int o = 64; o > 0; o >>= 1) { if (tid < o) red[tid] += red[tid + o]; __syncthreads(); }
    if (tid == 0) scores[s] = red[0] * (1.f / 32.f);   // 1/sqrt(DIM)
    __syncthreads();
  }

  float mx = -3.4e38f;
  for (int s = seg; s < S_; ++s) mx = fmaxf(mx, scores[s]);
  float w[S_]; float ssum = 0.f;
  for (int s = 0; s < S_; ++s) {
    w[s] = (s >= seg) ? __expf(scores[s] - mx) : 0.f;
    ssum += w[s];
  }
  const float inv = 1.f / ssum;

  for (int d = tid; d < DIM_; d += 128) {
    float acc = 0.f;
    for (int s = seg; s < S_; ++s) acc += w[s] * vr[((size_t)b * S_ + s) * DIM_ + d];
    float o = acc * inv;
    out[(size_t)bt * DIM_ + d] = o / (1.f + __expf(-o));   // silu
  }
}

// ---------------- host launcher ----------------
extern "C" void kernel_launch(void* const* d_in, const int* in_sizes, int n_in,
                              void* d_out, int out_size, void* d_ws, size_t ws_size,
                              hipStream_t stream) {
  const float* x        = (const float*)d_in[0];
  const float* ms       = (const float*)d_in[1];
  const float* W_rq     = (const float*)d_in[2];
  const float* W_rkv    = (const float*)d_in[3];
  const float* W_wq     = (const float*)d_in[4];
  const float* W_wk     = (const float*)d_in[5];
  const float* W_wv     = (const float*)d_in[6];
  const float* wmatter  = (const float*)d_in[7];

  float* out     = (float*)d_out;
  float* out_mem = out + (size_t)B_ * T_ * DIM_;        // memory tail of output

  // carve workspace
  char* w = (char*)d_ws;
  auto take = [&](size_t bytes) { char* p = w; w += (bytes + 255) & ~(size_t)255; return p; };
  const size_t NX = (size_t)GM * GK;                    // 16M
  const size_t NW = (size_t)GN * GK;                    // 1M
  __bf16* xb    = (__bf16*)take(NX * 2);
  __bf16* wbk   = (__bf16*)take(NW * 2);
  __bf16* wbv   = (__bf16*)take(NW * 2);
  __bf16* wbrq  = (__bf16*)take(NW * 2);
  __bf16* kb    = (__bf16*)take(NX * 2);
  __bf16* vb    = (__bf16*)take(NX * 2);
  __bf16* qrb   = (__bf16*)take(NX * 2);
  float*  qw    = (float*)take((size_t)S_ * DIM_ * 4);
  float*  memraw= (float*)take((size_t)B_ * S_ * DIM_ * 4);
  float*  kr    = (float*)take((size_t)B_ * S_ * DIM_ * 4);
  float*  vr    = (float*)take((size_t)B_ * S_ * DIM_ * 4);

  // 0: casts to bf16 (vectorized, 8 elements/thread)
  cast_f32_bf16<<<(int)(NX / 8 / 256), 256, 0, stream>>>(x, xb, (int)(NX / 8));
  cast_f32_bf16<<<(int)(NW / 8 / 256), 256, 0, stream>>>(W_wk, wbk, (int)(NW / 8));
  cast_f32_bf16<<<(int)(NW / 8 / 256), 256, 0, stream>>>(W_wv, wbv, (int)(NW / 8));
  cast_f32_bf16<<<(int)(NW / 8 / 256), 256, 0, stream>>>(W_rq, wbrq, (int)(NW / 8));

  // 1: three big WMMA GEMMs (16384 x 1024 x 1024 each), 64x64 tile per wave
  const int gemmBlocks = (GM / 64) * (GN / 64) / 8;     // 512
  wmma_gemm_xw<<<gemmBlocks, 256, 0, stream>>>(xb, wbk,  kb);
  wmma_gemm_xw<<<gemmBlocks, 256, 0, stream>>>(xb, wbv,  vb);
  wmma_gemm_xw<<<gemmBlocks, 256, 0, stream>>>(xb, wbrq, qrb);

  // 2: q = memory_slots @ W_write_q^T
  qw_gemm<<<(S_ * DIM_) / 256, 256, 0, stream>>>(ms, W_wq, qw);

  // 3: write attention
  write_attn<<<B_ * S_ * H_, 128, 0, stream>>>(qw, kb, vb, memraw);

  // 4: rms-norm -> memory (second output)
  mem_norm<<<B_ * S_, 256, 0, stream>>>(memraw, wmatter, out_mem);

  // 5: kr/vr = memory @ W_read_kv^T
  read_kv<<<(B_ * S_ * 2 * DIM_) / 256, 256, 0, stream>>>(out_mem, W_rkv, kr, vr);

  // 6: read attention + SiLU (first output)
  read_attn<<<B_ * T_, 128, 0, stream>>>(qrb, kr, vr, out);
}
